// SelfAttentionLayer_11416023073069
// MI455X (gfx1250) — matrix operand order repaired and checked by reading, not compiled, not run on hardware
//
#include <hip/hip_runtime.h>
#include <hip/hip_bf16.h>

// ---------------------------------------------------------------------------
// MI455X (gfx1250, wave32) fused self-attention with relative-attention bias.
//   - bf16 WMMA (v_wmma_f32_16x16x32_bf16) for all GEMMs, f32 accumulate
//   - GEMM: 16x64 tile per wave (A-fragment reuse, 2.5 b128 loads per WMMA)
//   - flash-attention fused kernel, 4 waves/WGP-workgroup sharing one 40KB
//     b_nd LDS copy; wave-local s_wait_dscnt ordering for the P transpose
// ---------------------------------------------------------------------------

typedef __bf16 bf16_t;
typedef __attribute__((ext_vector_type(16))) __bf16 v16bf;
typedef __attribute__((ext_vector_type(8)))  __bf16 v8bf;
typedef __attribute__((ext_vector_type(8)))  float  v8f;

// Problem constants (from reference)
constexpr int BATCH = 4;
constexpr int SEQ   = 1024;   // T == MAXLEN -> band is exactly causal
constexpr int EMB   = 1024;
constexpr int HEADS = 16;
constexpr int DQ    = 64;
constexpr int NB    = 10;
constexpr float NEGINF   = -1e9f;
constexpr float SM_SCALE = 0.125f;   // 1/sqrt(DQ)

#define DEVFN static __device__ __forceinline__

DEVFN v8f zero8() {
  v8f z;
#pragma unroll
  for (int i = 0; i < 8; ++i) z[i] = 0.f;
  return z;
}

// ---- WMMA fragment loaders (layouts per CDNA5 ISA §7.12.2) ------
// A operand, 16x32 bf16, row-major source [.. x lda]:
//   lane (hlf = lane>>4, m = lane&15): element e -> row m,
//   K = k0 + (e%8) + (e/8)*16 + hlf*8   => two contiguous 16B runs.
DEVFN v16bf frag_a_rm(const bf16_t* __restrict__ A, int lda, int row,
                      int k0, int hlf) {
  const bf16_t* p = A + (size_t)row * lda + k0 + hlf * 8;
  v8bf lo = *(const v8bf*)p;          // K = k0+hlf*8   .. +7   -> e 0..7
  v8bf hi = *(const v8bf*)(p + 16);   // K = k0+16+hlf*8.. +7   -> e 8..15
  v16bf r;
#pragma unroll
  for (int i = 0; i < 8; ++i) { r[i] = lo[i]; r[i + 8] = hi[i]; }
  return r;
}

// B operand, 32x16 bf16, loaded from a TRANSPOSED source Bt[n][k] (row-major,
// leading dim ldbt): element e -> k = k0 + (e%8) + (e/8)*16 + hlf*8, n = m.
// => two contiguous 16B runs (same k order as frag_a_rm).
DEVFN v16bf frag_b_bt(const bf16_t* __restrict__ Bt, int ldbt, int nrow,
                      int k0, int hlf) {
  const bf16_t* p = Bt + (size_t)nrow * ldbt + k0 + hlf * 8;
  v8bf lo = *(const v8bf*)p;
  v8bf hi = *(const v8bf*)(p + 16);
  v16bf r;
#pragma unroll
  for (int i = 0; i < 8; ++i) { r[i] = lo[i]; r[i + 8] = hi[i]; }
  return r;
}

DEVFN v8f wmma_bf16(v16bf a, v16bf b, v8f c) {
  // (neg_a, A, neg_b, B, c_mod, C, reuse_a, reuse_b)
  return __builtin_amdgcn_wmma_f32_16x16x32_bf16(
      false, a, false, b, (short)0, c, false, false);
}

// ---------------------------------------------------------------------------
// Elementwise convert / transpose-convert (prep kernels)
// ---------------------------------------------------------------------------
__global__ void cvt_f32_bf16(const float* __restrict__ in,
                             bf16_t* __restrict__ out, int n) {
  int i = blockIdx.x * blockDim.x + threadIdx.x;
  if (i < n) out[i] = (bf16_t)in[i];
}

// W [K x N] f32 row-major -> Wt [N x K] bf16 row-major
__global__ void transpose_f32_bf16(const float* __restrict__ W,
                                   bf16_t* __restrict__ Wt, int K, int N) {
  int i = blockIdx.x * blockDim.x + threadIdx.x;
  if (i < K * N) {
    int k = i / N, n = i % N;
    Wt[(size_t)n * K + k] = (bf16_t)W[i];
  }
}

// ---------------------------------------------------------------------------
// GEMM: C[M,N] = A[M,K](bf16) @ Bt[N,K](bf16)^T + bias, WMMA 16x16x32 bf16.
// 128 threads = 4 waves; each wave owns a 16x64 tile (one A frag feeds 4
// WMMAs per k-step). Nalloc = columns physically allocated in Bt (loads may
// touch [N, Nalloc) garbage; stores are guarded by col < N).
// out modes: 0 = f32 row-major [M,N]
//            1 = bf16 -> [B,H,T,DQ]   (Q, K)
//            2 = bf16 -> [B,H,DQ,T]   (V transposed for PV operand)
//            3 = f32  -> [B,H,T,NB]   (R)
// ---------------------------------------------------------------------------
__global__ __launch_bounds__(128) void gemm_wmma(
    const bf16_t* __restrict__ A, const bf16_t* __restrict__ Bt,
    const float* __restrict__ bias, void* __restrict__ C,
    int M, int N, int Nalloc, int K, int mode) {
  const int lane = threadIdx.x & 31;
  const int wave = threadIdx.x >> 5;
  const int m    = lane & 15;
  const int hlf  = lane >> 4;
  const int n0   = (blockIdx.x * 4 + wave) * 64;
  const int row0 = blockIdx.y * 16;
  if (n0 >= Nalloc) return;   // uniform per wave

  v8f acc[4] = {zero8(), zero8(), zero8(), zero8()};
  for (int k0 = 0; k0 < K; k0 += 32) {
    const v16bf a = frag_a_rm(A, K, row0 + m, k0, hlf);
#pragma unroll
    for (int s = 0; s < 4; ++s) {
      const v16bf b = frag_b_bt(Bt, K, n0 + s * 16 + m, k0, hlf);
      acc[s] = wmma_bf16(a, b, acc[s]);
    }
  }
#pragma unroll
  for (int s = 0; s < 4; ++s) {
    const int col = n0 + s * 16 + m;
    if (col >= N) continue;
    const float bv = bias ? bias[col] : 0.f;
#pragma unroll
    for (int r = 0; r < 8; ++r) {
      const int row = row0 + r + hlf * 8;  // C layout: VGPR r -> M = r + hlf*8
      const float v = acc[s][r] + bv;
      if (mode == 0) {
        ((float*)C)[(size_t)row * N + col] = v;
      } else if (mode == 1) {
        int b_ = row / SEQ, t = row % SEQ, h = col / DQ, d = col % DQ;
        ((bf16_t*)C)[(((size_t)b_ * HEADS + h) * SEQ + t) * DQ + d] = (bf16_t)v;
      } else if (mode == 2) {
        int b_ = row / SEQ, t = row % SEQ, h = col / DQ, d = col % DQ;
        ((bf16_t*)C)[(((size_t)b_ * HEADS + h) * DQ + d) * SEQ + t] = (bf16_t)v;
      } else {
        int b_ = row / SEQ, t = row % SEQ, h = col / NB, n = col % NB;
        ((float*)C)[(((size_t)b_ * HEADS + h) * SEQ + t) * NB + n] = v;
      }
    }
  }
}

// ---------------------------------------------------------------------------
// Fused flash attention with relattn bias.
// 128 threads = 4 waves (one per SIMD32 of the WGP); each wave owns one
// 16-query tile, all share one 40KB b_nd LDS copy. 32 keys per iteration:
// S (4 WMMA), online softmax + relattn bias in VALU from LDS, then P@V
// (4 WMMA, K=32) with P transposed through per-wave LDS. Wave trip counts
// differ (causal), so in-loop ordering is wave-local: same-wave DS ops are
// in-order (ISA ch.11) + explicit s_wait_dscnt before the transposed reads.
// ---------------------------------------------------------------------------
__global__ __launch_bounds__(128) void attn_wmma(
    const bf16_t* __restrict__ Q, const bf16_t* __restrict__ Kmat,
    const bf16_t* __restrict__ Vt, const float* __restrict__ R,
    const float* __restrict__ bnd, bf16_t* __restrict__ Aout) {
  __shared__ float bnd_lds[NB][SEQ];                    // 40 KB, shared by 4 waves
  __shared__ float R_lds[4][16][NB];                    // per-wave relattn queries
  __shared__ __align__(16) bf16_t P_lds[4][16][32];     // per-wave P transpose

  const int tid   = threadIdx.x;
  const int lane  = tid & 31;
  const int wave  = tid >> 5;
  const int m     = lane & 15;
  const int hlf   = lane >> 4;
  const int h     = blockIdx.y;
  const int b     = blockIdx.z;
  const int itile = blockIdx.x * 4 + wave;
  const int i0    = itile * 16;
  const size_t headoff = (size_t)b * HEADS + h;

  // Cooperative staging (uniform across the block), then one barrier.
  for (int i = tid; i < NB * SEQ; i += 128)
    ((float*)bnd_lds)[i] = bnd[i];
  {
    const float* rp = R + (headoff * SEQ + i0) * NB;
    for (int i = lane; i < 16 * NB; i += 32)
      ((float*)R_lds[wave])[i] = rp[i];
  }
  __syncthreads();

  const bf16_t* qb = Q    + (headoff * SEQ + i0) * DQ;
  const bf16_t* kb = Kmat +  headoff * SEQ * DQ;
  const bf16_t* vb = Vt   +  headoff * DQ * SEQ;   // [DQ][T]

  const v16bf qa0 = frag_a_rm(qb, DQ, m, 0,  hlf);
  const v16bf qa1 = frag_a_rm(qb, DQ, m, 32, hlf);

  v8f o0 = zero8(), o1 = zero8(), o2 = zero8(), o3 = zero8();
  float mrow[8], lrow[8];
#pragma unroll
  for (int r = 0; r < 8; ++r) { mrow[r] = -3.0e38f; lrow[r] = 0.f; }

  for (int j0 = 0; j0 <= i0; j0 += 32) {
    // Prefetch next key/value tiles into cache (global_prefetch_b8).
    if (j0 + 32 <= i0) {
      __builtin_prefetch(kb + (size_t)(j0 + 32 + lane) * DQ, 0, 1);
      __builtin_prefetch(vb + (size_t)(lane * 2) * SEQ + j0 + 32, 0, 1);
    }

    // ---- S = Q K^T (16 x 32 keys), f32 accum ----
    v8f s0 = zero8();
    s0 = wmma_bf16(qa0, frag_b_bt(kb, DQ, j0 + m,      0,  hlf), s0);
    s0 = wmma_bf16(qa1, frag_b_bt(kb, DQ, j0 + m,      32, hlf), s0);
    v8f s1 = zero8();
    s1 = wmma_bf16(qa0, frag_b_bt(kb, DQ, j0 + 16 + m, 0,  hlf), s1);
    s1 = wmma_bf16(qa1, frag_b_bt(kb, DQ, j0 + 16 + m, 32, hlf), s1);

    // ---- scale + causal mask + relattn bias + online softmax ----
    float p0[8], p1[8], alpha[8];
#pragma unroll
    for (int r = 0; r < 8; ++r) {
      const int row = r + hlf * 8;          // C layout row within tile
      const int qi  = i0 + row;
      const int d0  = qi - (j0 + m);        // relative distance, subtile 0
      const int d1  = d0 - 16;              // subtile 1
      const int d0c = d0 < 0 ? 0 : d0;
      const int d1c = d1 < 0 ? 0 : d1;
      float e0 = 0.f, e1 = 0.f;
#pragma unroll
      for (int n = 0; n < NB; ++n) {        // extra = R . b_nd[:, d]
        const float rv = R_lds[wave][row][n];
        e0 += rv * bnd_lds[n][d0c];
        e1 += rv * bnd_lds[n][d1c];
      }
      const float x0 = (d0 >= 0) ? s0[r] * SM_SCALE + e0 : NEGINF;
      const float x1 = (d1 >= 0) ? s1[r] * SM_SCALE + e1 : NEGINF;

      float mx = fmaxf(x0, x1);             // row lives across a 16-lane group
#pragma unroll
      for (int off = 8; off >= 1; off >>= 1)
        mx = fmaxf(mx, __shfl_xor(mx, off, 32));
      const float mn  = fmaxf(mrow[r], mx);
      const float al  = __expf(mrow[r] - mn);
      const float pv0 = __expf(x0 - mn);
      const float pv1 = __expf(x1 - mn);
      float rs = pv0 + pv1;
#pragma unroll
      for (int off = 8; off >= 1; off >>= 1)
        rs += __shfl_xor(rs, off, 32);
      lrow[r]  = lrow[r] * al + rs;
      mrow[r]  = mn;
      alpha[r] = al;
      p0[r] = pv0;
      p1[r] = pv1;
    }
#pragma unroll
    for (int r = 0; r < 8; ++r) {
      o0[r] *= alpha[r]; o1[r] *= alpha[r];
      o2[r] *= alpha[r]; o3[r] *= alpha[r];
    }

    // ---- transpose P (D-layout -> A-layout) through wave-private LDS ----
    // Compiler barrier: keep these stores after the previous iteration's
    // reads; hardware keeps same-wave DS ops in order.
    asm volatile("" ::: "memory");
#pragma unroll
    for (int r = 0; r < 8; ++r) {
      P_lds[wave][r + hlf * 8][m]      = (bf16_t)p0[r];
      P_lds[wave][r + hlf * 8][16 + m] = (bf16_t)p1[r];
    }
    // Wave-local DS drain before the cross-lane transposed reads.
    asm volatile("s_wait_dscnt 0x0" ::: "memory");
    const v16bf pa = frag_a_rm(&P_lds[wave][0][0], 32, m, 0, hlf);

    // ---- O += P @ V (V stored [DQ][T] so B-operand is contiguous) ----
    o0 = wmma_bf16(pa, frag_b_bt(vb, SEQ, 0  + m, j0, hlf), o0);
    o1 = wmma_bf16(pa, frag_b_bt(vb, SEQ, 16 + m, j0, hlf), o1);
    o2 = wmma_bf16(pa, frag_b_bt(vb, SEQ, 32 + m, j0, hlf), o2);
    o3 = wmma_bf16(pa, frag_b_bt(vb, SEQ, 48 + m, j0, hlf), o3);
  }

  // ---- epilogue: normalize and write bf16 [B, T, H*DQ] for final GEMM ----
#pragma unroll
  for (int r = 0; r < 8; ++r) {
    const int row = r + hlf * 8;
    const int t   = i0 + row;
    const float inv = 1.0f / lrow[r];
    bf16_t* op = Aout + ((size_t)b * SEQ + t) * (HEADS * DQ) + (size_t)h * DQ;
    op[0  + m] = (bf16_t)(o0[r] * inv);
    op[16 + m] = (bf16_t)(o1[r] * inv);
    op[32 + m] = (bf16_t)(o2[r] * inv);
    op[48 + m] = (bf16_t)(o3[r] * inv);
  }
}

// ---------------------------------------------------------------------------
// Host-side orchestration
// ---------------------------------------------------------------------------
extern "C" void kernel_launch(void* const* d_in, const int* in_sizes, int n_in,
                              void* d_out, int out_size, void* d_ws,
                              size_t ws_size, hipStream_t stream) {
  (void)in_sizes; (void)n_in; (void)out_size; (void)ws_size;
  const float* X     = (const float*)d_in[0];
  const float* Wq    = (const float*)d_in[1];
  const float* bq    = (const float*)d_in[2];
  const float* Wk    = (const float*)d_in[3];
  const float* Wv    = (const float*)d_in[4];
  const float* Wproj = (const float*)d_in[5];
  const float* bproj = (const float*)d_in[6];
  const float* Wr    = (const float*)d_in[7];
  const float* br    = (const float*)d_in[8];
  const float* bnd   = (const float*)d_in[9];
  float* out = (float*)d_out;

  constexpr int MT = BATCH * SEQ;        // 4096 rows
  constexpr int HD = HEADS * DQ;         // 1024
  constexpr int HN = HEADS * NB;         // 160
  constexpr int HN_PAD = 192;            // R weight cols padded to 64-multiple

  // Workspace carve-out (~51 MiB), 256B aligned slices.
  char* ws = (char*)d_ws;
  auto carve = [&](size_t bytes) {
    char* p = ws;
    ws += (bytes + 255) & ~(size_t)255;
    return p;
  };
  bf16_t* Xb  = (bf16_t*)carve((size_t)MT * EMB * 2);
  bf16_t* Wqt = (bf16_t*)carve((size_t)EMB * HD * 2);
  bf16_t* Wkt = (bf16_t*)carve((size_t)EMB * HD * 2);
  bf16_t* Wvt = (bf16_t*)carve((size_t)EMB * HD * 2);
  bf16_t* Wrt = (bf16_t*)carve((size_t)EMB * HN_PAD * 2);  // rows [160,192) garbage, never stored
  bf16_t* Wpt = (bf16_t*)carve((size_t)HD * EMB * 2);
  bf16_t* Qb  = (bf16_t*)carve((size_t)MT * HD * 2);   // [B,H,T,DQ]
  bf16_t* Kb  = (bf16_t*)carve((size_t)MT * HD * 2);   // [B,H,T,DQ]
  bf16_t* Vtb = (bf16_t*)carve((size_t)MT * HD * 2);   // [B,H,DQ,T]
  float*  Rf  = (float*) carve((size_t)MT * HN * 4);   // [B,H,T,NB]
  bf16_t* Ao  = (bf16_t*)carve((size_t)MT * HD * 2);   // [B,T,H*DQ]

  // 1) precision conversion + weight transposes
  {
    int n = MT * EMB;
    cvt_f32_bf16<<<(n + 255) / 256, 256, 0, stream>>>(X, Xb, n);
  }
  {
    int n = EMB * HD;
    transpose_f32_bf16<<<(n + 255) / 256, 256, 0, stream>>>(Wq, Wqt, EMB, HD);
    transpose_f32_bf16<<<(n + 255) / 256, 256, 0, stream>>>(Wk, Wkt, EMB, HD);
    transpose_f32_bf16<<<(n + 255) / 256, 256, 0, stream>>>(Wv, Wvt, EMB, HD);
    transpose_f32_bf16<<<(n + 255) / 256, 256, 0, stream>>>(Wproj, Wpt, HD, EMB);
  }
  {
    int n = EMB * HN;
    transpose_f32_bf16<<<(n + 255) / 256, 256, 0, stream>>>(Wr, Wrt, EMB, HN);
  }

  // 2) projection GEMMs (WMMA bf16, 16x64 tile per wave)
  dim3 blk(128);
  dim3 grid_n1024(HD / 256, MT / 16);    // (4, 256)
  dim3 grid_n160(1, MT / 16);            // waves 0..2 active (Nalloc=192)
  gemm_wmma<<<grid_n1024, blk, 0, stream>>>(Xb, Wqt, bq,      Qb,  MT, HD, HD,     EMB, 1);
  gemm_wmma<<<grid_n1024, blk, 0, stream>>>(Xb, Wkt, nullptr, Kb,  MT, HD, HD,     EMB, 1);
  gemm_wmma<<<grid_n1024, blk, 0, stream>>>(Xb, Wvt, nullptr, Vtb, MT, HD, HD,     EMB, 2);
  gemm_wmma<<<grid_n160,  blk, 0, stream>>>(Xb, Wrt, br,      Rf,  MT, HN, HN_PAD, EMB, 3);

  // 3) fused flash attention with relattn bias (4 waves per block)
  dim3 agrid(SEQ / 64, HEADS, BATCH);    // 4 query tiles per block
  attn_wmma<<<agrid, 128, 0, stream>>>(Qb, Kb, Vtb, Rf, bnd, Ao);

  // 4) output projection -> d_out (f32, [B,T,E])
  gemm_wmma<<<grid_n1024, blk, 0, stream>>>(Ao, Wpt, bproj, out, MT, EMB, EMB, HD, 0);
}
